// CleanAttention_37915971289913
// MI455X (gfx1250) — compile-verified, matrix operand order checked by reading, not run on hardware
//
#include <hip/hip_runtime.h>
#include <hip/hip_bf16.h>
#include <math.h>
#include <stdint.h>

#define DEVI __device__ __forceinline__

typedef __attribute__((ext_vector_type(16))) _Float16 v16h;
typedef __attribute__((ext_vector_type(8)))  _Float16 v8h;
typedef __attribute__((ext_vector_type(8)))  float    v8f;
typedef __attribute__((ext_vector_type(4)))  uint32_t v4u;
typedef __attribute__((ext_vector_type(8)))  uint32_t v8u;

// ---- problem constants (from reference) ----
constexpr int BATCH = 2, SEQ = 4096, CDIM = 2048;
constexpr int NH = 16, NKV = 4, HD = 128;
constexpr int WIN = 256;
constexpr int MROWS = BATCH * SEQ;            // 8192
constexpr float SM_SCALE = 0.08838834764831845f; // 1/sqrt(128)

// ---- WMMA helpers ----
DEVI v8f vzero8() { v8f z; for (int i = 0; i < 8; ++i) z[i] = 0.f; return z; }

DEVI v8f wmma_f16(v16h a, v16h b, v8f c) {
  return __builtin_amdgcn_wmma_f32_16x16x32_f16(false, a, false, b,
                                                (short)0, c, false, false);
}

// Load a 16x32 (f16) operand fragment from an LDS tile laid out row-major
// with leading dimension `ld` (halves), per ISA 7.12.2:
//   lanes 0-15  : row = lane,     K = k0 + {0..7, 16..23}
//   lanes 16-31 : row = lane-16,  K = k0 + {8..15, 24..31}
DEVI v16h ld_frag(const _Float16* tile, int ld, int row0, int k0) {
  int lane = threadIdx.x & 31;
  const _Float16* p = tile + (size_t)(row0 + (lane & 15)) * ld + k0 + ((lane >> 4) << 3);
  union { v16h v; v8h h[2]; } u;
  u.h[0] = *(const v8h*)(p);
  u.h[1] = *(const v8h*)(p + 16);
  return u.v;
}

// ---- CDNA5 async copy: one 16-byte global->LDS transfer per lane ----
DEVI void async_cp16(const _Float16* g, _Float16* l) {
  asm volatile("global_load_async_to_lds_b128 %0, %1, off"
               :: "v"((uint32_t)(uintptr_t)l), "v"((uint64_t)(uintptr_t)g)
               : "memory");
}
DEVI void async_wait0() { asm volatile("s_wait_asynccnt 0x0" ::: "memory"); }

// ---- CDNA5 Tensor Data Mover: 128x32 f16 tile, row stride CDIM halves,
//      LDS rows padded to 40 halves (tile row = 16 DWORDs + 4 DWORDs pad) ----
DEVI void tdm_load_tile_128x32(const _Float16* gsrc, _Float16* ldst) {
  uint64_t ga = (uint64_t)(uintptr_t)gsrc;
  uint32_t la = (uint32_t)(uintptr_t)ldst;
  v4u g0;
  g0[0] = 1u;                                    // count=1 (valid), user mode
  g0[1] = la;                                    // lds_addr (bytes)
  g0[2] = (uint32_t)ga;                          // global_addr[31:0]
  g0[3] = ((uint32_t)(ga >> 32) & 0x01FFFFFFu)   // global_addr[56:32]
          | (2u << 30);                          // type=2 ("image")
  v8u g1;
  g1[0] = (1u << 16)      // data_size = 2 bytes
        | (1u << 20)      // pad_enable
        | (3u << 22)      // pad_interval: 16 DWORDs between pads
        | (3u << 25);     // pad_amount: 4 DWORDs (8 halves) -> LDT = 40
  g1[1] = (uint32_t)CDIM << 16;                  // tensor_dim0[15:0]
  g1[2] = (uint32_t)8192u << 16;                 // dim0[31:16]=0 | tensor_dim1[15:0]
  g1[3] = 32u << 16;                             // dim1[31:16]=0 | tile_dim0 = 32
  g1[4] = 128u;                                  // tile_dim1 = 128 | tile_dim2 = 0
  g1[5] = (uint32_t)CDIM;                        // tensor_dim0_stride[31:0]
  g1[6] = 0u;                                    // stride[47:32] | dim1_stride[15:0]
  g1[7] = 0u;
  asm volatile("tensor_load_to_lds %0, %1" :: "s"(g0), "s"(g1) : "memory");
}

// ---- f32 -> f16 conversion (n divisible by 4) ----
__global__ void cvt_f32_f16(const float* __restrict__ src,
                            _Float16* __restrict__ dst, int n) {
  int i = (blockIdx.x * 256 + threadIdx.x) * 4;
  if (i + 3 < n) {
    float4 v = *(const float4*)(src + i);
    dst[i + 0] = (_Float16)v.x;
    dst[i + 1] = (_Float16)v.y;
    dst[i + 2] = (_Float16)v.z;
    dst[i + 3] = (_Float16)v.w;
  }
}

// ---- tiled WMMA GEMM (TDM double-buffered): D[m,n] = sum_k A[m,k]*Wt[n,k] ----
// MODE 0: Q proj -> RoPE -> qh[B,NH,T,D]   MODE 1: K proj -> RoPE -> kh[B,NKV,T,D]
// MODE 2: V proj ->         vh[B,NKV,T,D]  MODE 3: out proj ->      d_out (f32)
constexpr int GK  = 32;        // K tile (halves)
constexpr int LDT = GK + 8;    // padded LDS stride (matches TDM pad config)
constexpr int NIT = CDIM / GK; // 64 K-iterations
constexpr int GEMM_SMEM = 128 * 132 * 4;  // f32 stage dominates 4 x (128*LDT) tiles

template <int MODE>
__global__ __launch_bounds__(256)
void gemm_k(const _Float16* __restrict__ A, const _Float16* __restrict__ Wt,
            const float* __restrict__ rope, void* __restrict__ dstp, int Ntot) {
  (void)Ntot;
  extern __shared__ char smem[];
  _Float16* tiles = (_Float16*)smem;       // [2 bufs][A|B][128][LDT]
  float* stage = (float*)smem;             // aliased after K-loop: [128][132]

  const int tid  = threadIdx.x;
  const int lane = tid & 31;
  const int w    = tid >> 5;               // 8 waves
  const int wm   = w >> 1;                 // 0..3 -> rows wm*32
  const int wn   = w & 1;                  // 0..1 -> cols wn*64
  const int m0   = blockIdx.y * 128;
  const int n0   = blockIdx.x * 128;

  const _Float16* Ag = A  + (size_t)m0 * CDIM;
  const _Float16* Bg = Wt + (size_t)n0 * CDIM;

  v8f acc[2][4];
  for (int i = 0; i < 2; ++i)
    for (int j = 0; j < 4; ++j) acc[i][j] = vzero8();

  // prime buffer 0 via Tensor Data Mover (wave 0 issues; EXEC-independent DMA)
  if (w == 0) {
    tdm_load_tile_128x32(Ag, tiles + 0 * 128 * LDT);
    tdm_load_tile_128x32(Bg, tiles + 1 * 128 * LDT);
  }

  for (int it = 0; it < NIT; ++it) {
    const int buf = it & 1;
    if (w == 0) {
      if (it + 1 < NIT) {   // issue next tile pair into the other buffer
        const int nk = (it + 1) * GK;
        tdm_load_tile_128x32(Ag + nk, tiles + ((buf ^ 1) * 2 + 0) * 128 * LDT);
        tdm_load_tile_128x32(Bg + nk, tiles + ((buf ^ 1) * 2 + 1) * 128 * LDT);
        __builtin_amdgcn_s_wait_tensorcnt(2);  // current pair done (in-order)
      } else {
        __builtin_amdgcn_s_wait_tensorcnt(0);
      }
    }
    __syncthreads();

    const _Float16* As = tiles + (buf * 2 + 0) * 128 * LDT;
    const _Float16* Bs = tiles + (buf * 2 + 1) * 128 * LDT;
    v16h bfrag[4];
    for (int ni = 0; ni < 4; ++ni)
      bfrag[ni] = ld_frag(Bs, LDT, wn * 64 + ni * 16, 0);
    for (int mi = 0; mi < 2; ++mi) {
      v16h afrag = ld_frag(As, LDT, wm * 32 + mi * 16, 0);
      for (int ni = 0; ni < 4; ++ni)
        acc[mi][ni] = wmma_f16(afrag, bfrag[ni], acc[mi][ni]);
    }
    __syncthreads();
  }

  // stage accumulators (f32) to LDS for cross-lane epilogue
  {
    int half8 = (lane >> 4) << 3;
    for (int mi = 0; mi < 2; ++mi)
      for (int ni = 0; ni < 4; ++ni)
        for (int r = 0; r < 8; ++r) {
          int row = wm * 32 + mi * 16 + r + half8;
          int col = wn * 64 + ni * 16 + (lane & 15);
          stage[row * 132 + col] = acc[mi][ni][r];
        }
  }
  __syncthreads();

  if (MODE == 3) {
    float* out = (float*)dstp;
    for (int p = tid; p < 128 * 128; p += 256) {
      int row = p >> 7, col = p & 127;
      out[(size_t)(m0 + row) * CDIM + (n0 + col)] = stage[row * 132 + col];
    }
  } else if (MODE == 2) {
    _Float16* vh = (_Float16*)dstp;
    for (int p = tid; p < 128 * 128; p += 256) {
      int row = p >> 7, col = p & 127;
      int gr = m0 + row, gc = n0 + col;
      int b = gr / SEQ, t = gr % SEQ;
      int h = gc >> 7, dd = gc & 127;
      vh[(((size_t)(b * NKV + h) * SEQ) + t) * HD + dd] =
          (_Float16)stage[row * 132 + col];
    }
  } else {  // MODE 0/1: RoPE over (even, odd) pairs
    const int H = (MODE == 0) ? NH : NKV;
    _Float16* qk = (_Float16*)dstp;
    for (int p = tid; p < 128 * 64; p += 256) {
      int row = p >> 6, i = p & 63;
      int gr = m0 + row, gc = n0 + 2 * i;
      int b = gr / SEQ, t = gr % SEQ;
      int h = gc >> 7, dd = gc & 127;        // dd even
      const float* rc = rope + ((size_t)t * 64 + (dd >> 1)) * 2;
      float cs = rc[0], sn = rc[1];
      float x0 = stage[row * 132 + 2 * i];
      float x1 = stage[row * 132 + 2 * i + 1];
      size_t base = (((size_t)(b * H + h) * SEQ) + t) * HD + dd;
      qk[base]     = (_Float16)(x0 * cs - x1 * sn);
      qk[base + 1] = (_Float16)(x0 * sn + x1 * cs);
    }
  }
}

// ---- sliding-window attention (flash-style, 128-query chunks, 3 key slabs) ----
constexpr int LDK  = HD + 8;    // 136 halves
constexpr int SROW = 132;       // f32 score-row stride
constexpr int ATT_SMEM = 3 * 128 * LDK * 2 + 128 * SROW * 4 + 3 * 128 * 4;

__global__ __launch_bounds__(256)
void attn_k(const _Float16* __restrict__ qh, const _Float16* __restrict__ kh,
            const _Float16* __restrict__ vh, _Float16* __restrict__ yh) {
  extern __shared__ char smem[];
  _Float16* Qs = (_Float16*)smem;          // [128 query][LDK]
  _Float16* Ks = Qs + 128 * LDK;           // [128 key][LDK]
  _Float16* Vt = Ks + 128 * LDK;           // [128 d][LDK]  (V transposed)
  float* Sf   = (float*)(Vt + 128 * LDK);  // [128][SROW] scores; P aliases rows
  float* mrow = Sf + 128 * SROW;
  float* lrow = mrow + 128;
  float* srow = lrow + 128;

  const int tid = threadIdx.x, lane = tid & 31, w = tid >> 5;
  const int c = blockIdx.x;                // query chunk 0..31
  const int h = blockIdx.y;                // head
  const int b = blockIdx.z;
  const int hk = h >> 2;                   // GQA: 4 query heads per kv head
  const int t0 = c * 128;

  const _Float16* qbase  = qh + (((size_t)(b * NH + h) * SEQ) + t0) * HD;
  const _Float16* kbase0 = kh + ((size_t)(b * NKV + hk) * SEQ) * HD;
  const _Float16* vbase0 = vh + ((size_t)(b * NKV + hk) * SEQ) * HD;

  // Q tile: async direct-to-LDS copies (ASYNCcnt)
  for (int p = tid; p < 128 * 16; p += 256) {
    int row = p / 16, col8 = (p % 16) * 8;
    async_cp16(qbase + (size_t)row * HD + col8, Qs + row * LDK + col8);
  }
  if (tid < 128) { mrow[tid] = -1e30f; lrow[tid] = 0.f; }

  v8f acc[8];
  for (int i = 0; i < 8; ++i) acc[i] = vzero8();
  async_wait0();
  __syncthreads();

  for (int s = 0; s < 3; ++s) {
    int kb = c - 2 + s;
    if (kb < 0) continue;
    int kt0 = kb * 128;

    for (int p = tid; p < 128 * 16; p += 256) {     // K slab: async -> LDS
      int row = p / 16, col8 = (p % 16) * 8;
      async_cp16(kbase0 + (size_t)(kt0 + row) * HD + col8,
                 Ks + row * LDK + col8);
    }
    for (int p = tid; p < 128 * 128; p += 256) {    // V slab, transposed
      int key = p >> 7, d = p & 127;
      Vt[d * LDK + key] = vbase0[(size_t)(kt0 + key) * HD + d];
    }
    async_wait0();
    __syncthreads();

    // S = (Q K^T) * scale   (wave w owns query rows w*16..w*16+15)
    for (int nt = 0; nt < 8; ++nt) {
      v8f sfr = vzero8();
      for (int ks = 0; ks < 4; ++ks) {
        v16h a  = ld_frag(Qs, LDK, w * 16, ks * 32);
        v16h bf = ld_frag(Ks, LDK, nt * 16, ks * 32);
        sfr = wmma_f16(a, bf, sfr);
      }
      int rowb = w * 16 + ((lane >> 4) << 3);
      int col  = nt * 16 + (lane & 15);
      for (int r = 0; r < 8; ++r)
        Sf[(rowb + r) * SROW + col] = sfr[r] * SM_SCALE;
    }
    __syncthreads();

    // online softmax, one thread per query row
    if (tid < 128) {
      int row = tid, tq = t0 + row;
      float mOld = mrow[row];
      float rmax = -1e30f;
      for (int k = 0; k < 128; ++k) {
        int tk = kt0 + k;
        if (tk <= tq && (tq - tk) < WIN)
          rmax = fmaxf(rmax, Sf[row * SROW + k]);
      }
      float mNew = fmaxf(mOld, rmax);
      float corr = __expf(mOld - mNew);     // finite sentinel: exp(0)=1 when empty
      _Float16* prow = (_Float16*)(Sf + row * SROW);  // in-place P (alias-safe)
      float lsum = 0.f;
      for (int k = 0; k < 128; ++k) {
        int tk = kt0 + k;
        float pv = 0.f;
        if (tk <= tq && (tq - tk) < WIN)
          pv = __expf(Sf[row * SROW + k] - mNew);
        prow[k] = (_Float16)pv;
        lsum += pv;
      }
      mrow[row] = mNew;
      lrow[row] = lrow[row] * corr + lsum;
      srow[row] = corr;
    }
    __syncthreads();

    // rescale accumulators, then O += P * V
    {
      int half8 = (lane >> 4) << 3;
      float f[8];
      for (int r = 0; r < 8; ++r) f[r] = srow[w * 16 + r + half8];
      for (int nt = 0; nt < 8; ++nt)
        for (int r = 0; r < 8; ++r) acc[nt][r] *= f[r];
    }
    const _Float16* Ph = (const _Float16*)Sf;   // rows at stride 2*SROW halves
    for (int nt = 0; nt < 8; ++nt) {
      for (int ks = 0; ks < 4; ++ks) {
        v16h a  = ld_frag(Ph, 2 * SROW, w * 16, ks * 32);
        v16h bf = ld_frag(Vt, LDK, nt * 16, ks * 32);
        acc[nt] = wmma_f16(a, bf, acc[nt]);
      }
    }
    __syncthreads();
  }

  // normalize and write y[b, t, h*D + d] (f16, row-major C)
  {
    int half8 = (lane >> 4) << 3;
    for (int nt = 0; nt < 8; ++nt)
      for (int r = 0; r < 8; ++r) {
        int row = w * 16 + r + half8;
        float l = lrow[row];
        float val = (l > 0.f) ? acc[nt][r] / l : 0.f;
        int tq = t0 + row;
        yh[((size_t)b * SEQ + tq) * CDIM + h * HD + nt * 16 + (lane & 15)] =
            (_Float16)val;
      }
  }
}

// ---- host ----
extern "C" void kernel_launch(void* const* d_in, const int* in_sizes, int n_in,
                              void* d_out, int out_size, void* d_ws, size_t ws_size,
                              hipStream_t stream) {
  (void)in_sizes; (void)n_in; (void)out_size; (void)ws_size;
  const float* x    = (const float*)d_in[0];
  const float* wq   = (const float*)d_in[1];
  const float* wk   = (const float*)d_in[2];
  const float* wv   = (const float*)d_in[3];
  const float* wo   = (const float*)d_in[4];
  const float* rope = (const float*)d_in[5];

  _Float16* ws = (_Float16*)d_ws;
  size_t o = 0;
  _Float16* xh  = ws + o; o += (size_t)MROWS * CDIM;
  _Float16* wqh = ws + o; o += (size_t)(NH  * HD) * CDIM;
  _Float16* wkh = ws + o; o += (size_t)(NKV * HD) * CDIM;
  _Float16* wvh = ws + o; o += (size_t)(NKV * HD) * CDIM;
  _Float16* woh = ws + o; o += (size_t)CDIM * (NH * HD);
  _Float16* qh  = ws + o; o += (size_t)BATCH * NH  * SEQ * HD;
  _Float16* kh  = ws + o; o += (size_t)BATCH * NKV * SEQ * HD;
  _Float16* vh  = ws + o; o += (size_t)BATCH * NKV * SEQ * HD;
  _Float16* yh  = ws + o; o += (size_t)MROWS * CDIM;

  auto cvt = [&](const float* s, _Float16* d, size_t n) {
    cvt_f32_f16<<<(unsigned)(n / 4 / 256), 256, 0, stream>>>(s, d, (int)n);
  };
  cvt(x,  xh,  (size_t)MROWS * CDIM);
  cvt(wq, wqh, (size_t)(NH  * HD) * CDIM);
  cvt(wk, wkh, (size_t)(NKV * HD) * CDIM);
  cvt(wv, wvh, (size_t)(NKV * HD) * CDIM);
  cvt(wo, woh, (size_t)CDIM * (NH * HD));

  dim3 blk(256);
  gemm_k<0><<<dim3(16, 64), blk, GEMM_SMEM, stream>>>(xh, wqh, rope, qh, NH * HD);
  gemm_k<1><<<dim3(4,  64), blk, GEMM_SMEM, stream>>>(xh, wkh, rope, kh, NKV * HD);
  gemm_k<2><<<dim3(4,  64), blk, GEMM_SMEM, stream>>>(xh, wvh, nullptr, vh, NKV * HD);
  attn_k<<<dim3(SEQ / 128, NH, BATCH), blk, ATT_SMEM, stream>>>(qh, kh, vh, yh);
  gemm_k<3><<<dim3(16, 64), blk, GEMM_SMEM, stream>>>(yh, woh, nullptr, d_out, CDIM);
}
